// GPTDecoder_57818849738820
// MI455X (gfx1250) — compile-verified, matrix operand order checked by reading, not run on hardware
//
#include <hip/hip_runtime.h>
#include <math.h>

// ---- model constants (match reference) ----
static constexpr int Bc  = 2;
static constexpr int Tc  = 1024;
static constexpr int Dc  = 1024;
static constexpr int Hc  = 16;
static constexpr int HDc = 64;     // Dc / Hc
static constexpr int DFFc= 4096;
static constexpr int Lc  = 8;
static constexpr int Vc  = 32000;
static constexpr int Mc  = Bc * Tc;   // 2048 rows

typedef __attribute__((ext_vector_type(16))) _Float16 v16h;
typedef __attribute__((ext_vector_type(8)))  _Float16 v8h;
typedef __attribute__((ext_vector_type(8)))  float    v8f;
typedef __attribute__((ext_vector_type(4)))  int      v4i;

// Async global->LDS copy path (CDNA5 ASYNCcnt pipe), guarded so the file
// compiles on toolchains where the builtins are absent.
#if defined(__AMDGCN__) &&                                            \
    __has_builtin(__builtin_amdgcn_global_load_async_to_lds_b128) &&  \
    __has_builtin(__builtin_amdgcn_s_wait_asynccnt)
#define USE_ASYNC_LDS 1
#else
#define USE_ASYNC_LDS 0
#endif

__device__ inline v8f wmma_f16(v16h a, v16h b, v8f c) {
  // (neg_a, A, neg_b, B, c_mod, C, reuse_a, reuse_b)
  return __builtin_amdgcn_wmma_f32_16x16x32_f16(false, a, false, b, (short)0, c,
                                                false, false);
}

// ---------------------------------------------------------------------------
// Embedding: h[bt, :] = tok_emb[x[bt], :] + pos_emb[bt % T, :]
// ---------------------------------------------------------------------------
__global__ __launch_bounds__(256)
void embed_kernel(const int* __restrict__ x, const float* __restrict__ tok,
                  const float* __restrict__ pos, float* __restrict__ h) {
  int bt = blockIdx.x;
  int t  = bt % Tc;
  int v  = x[bt];
  for (int i = threadIdx.x; i < Dc; i += 256)
    h[(size_t)bt * Dc + i] = tok[(size_t)v * Dc + i] + pos[(size_t)t * Dc + i];
}

// ---------------------------------------------------------------------------
// LayerNorm over last dim (Dn), fp32 in -> f16 out
// ---------------------------------------------------------------------------
__global__ __launch_bounds__(256)
void layernorm_kernel(const float* __restrict__ X, const float* __restrict__ g,
                      const float* __restrict__ bta, _Float16* __restrict__ Y,
                      int Dn) {
  int row = blockIdx.x;
  const float* x = X + (size_t)row * Dn;
  __shared__ float red[256];
  float s = 0.f;
  for (int i = threadIdx.x; i < Dn; i += 256) s += x[i];
  red[threadIdx.x] = s; __syncthreads();
  for (int off = 128; off > 0; off >>= 1) {
    if (threadIdx.x < off) red[threadIdx.x] += red[threadIdx.x + off];
    __syncthreads();
  }
  float mean = red[0] / Dn;
  __syncthreads();
  float vs = 0.f;
  for (int i = threadIdx.x; i < Dn; i += 256) { float d = x[i] - mean; vs += d * d; }
  red[threadIdx.x] = vs; __syncthreads();
  for (int off = 128; off > 0; off >>= 1) {
    if (threadIdx.x < off) red[threadIdx.x] += red[threadIdx.x + off];
    __syncthreads();
  }
  float rstd = rsqrtf(red[0] / Dn + 1e-5f);
  for (int i = threadIdx.x; i < Dn; i += 256)
    Y[(size_t)row * Dn + i] = (_Float16)((x[i] - mean) * rstd * g[i] + bta[i]);
}

// ---------------------------------------------------------------------------
// Tiled WMMA GEMM:  C(MxN) = A(MxK, f16) @ W(KxN, f32 -> f16 inline)
// Block tile 64x64, 256 threads = 8 waves (4x2), each wave 16x32 of C.
// A tile staged via async global->LDS (ASYNCcnt) when available; W tile is
// converted fp32->f16 through VGPRs (overlaps with the async copy).
// Epilogue: optional bias, ReLU, fp32 residual accumulate, f16/f32 store.
// All of M,N divisible by 64; K divisible by 32.
// ---------------------------------------------------------------------------
#define TM 64
#define TN 64
#define TK 32

__global__ __launch_bounds__(256)
void gemm_wmma_kernel(const _Float16* __restrict__ A, const float* __restrict__ W,
                      const float* __restrict__ bias,
                      float* __restrict__ Cres,      // fp32 residual accumulate (or null)
                      _Float16* __restrict__ Oh,     // f16 store (or null)
                      float* __restrict__ Of,        // fp32 store (or null)
                      int N, int K, int relu) {
  __shared__ alignas(64) _Float16 Alds[TM * TK];     // row-major, K contiguous
  __shared__ alignas(64) _Float16 Wlds[TN * TK];     // col-major, K contiguous

  const int m0 = blockIdx.y * TM;
  const int n0 = blockIdx.x * TN;
  const int tid  = threadIdx.x;
  const int lane = tid & 31;
  const int wave = tid >> 5;
  const int wm = wave & 3;      // 0..3  -> 16-row strip
  const int wn = wave >> 2;     // 0..1  -> 32-col strip
  const int lh = lane >> 4;     // lane half (K sub-range select)
  const int ln = lane & 15;

  v8f acc0 = {}, acc1 = {};

  const int arow = tid >> 2, acg = (tid & 3) * 8;    // A stage: 64 rows x 32 cols
  const int wkk  = tid & 31, wnb = (tid >> 5) * 8;   // W stage: 32 k  x 64 n

  for (int k0 = 0; k0 < K; k0 += TK) {
    // ---- Stage A tile (f16): async copy straight into LDS if available ----
#if USE_ASYNC_LDS
    __builtin_amdgcn_global_load_async_to_lds_b128(
        (__attribute__((address_space(1))) v4i*)&A[(size_t)(m0 + arow) * K + k0 + acg],
        (__attribute__((address_space(3))) v4i*)&Alds[arow * TK + acg],
        0, 0);
#else
    *(v8h*)&Alds[arow * TK + acg] =
        *(const v8h*)&A[(size_t)(m0 + arow) * K + k0 + acg];
#endif
    // ---- Stage W tile: fp32 load, convert, transpose into col-major f16 ----
    v8f wv = *(const v8f*)&W[(size_t)(k0 + wkk) * N + n0 + wnb];
#pragma unroll
    for (int i = 0; i < 8; ++i)
      Wlds[(wnb + i) * TK + wkk] = (_Float16)wv[i];
    if (k0 + TK < K)
      __builtin_prefetch(&W[(size_t)(k0 + TK + wkk) * N + n0 + wnb], 0, 0);
#if USE_ASYNC_LDS
    __builtin_amdgcn_s_wait_asynccnt(0);   // A tile resident before barrier
#endif
    __syncthreads();

    // A operand: lane ln holds row, lane half selects K 0..15 / 16..31
    v16h a  = *(const v16h*)&Alds[(wm * 16 + ln) * TK + lh * 16];
    v16h b0 = *(const v16h*)&Wlds[(wn * 32 + ln) * TK + lh * 16];
    v16h b1 = *(const v16h*)&Wlds[(wn * 32 + 16 + ln) * TK + lh * 16];
    acc0 = wmma_f16(a, b0, acc0);
    acc1 = wmma_f16(a, b1, acc1);
    __syncthreads();
  }

  // Epilogue. C/D layout: vgpr r, lanes 0-15 -> M=r ; lanes 16-31 -> M=8+r.
  const int mbase = m0 + wm * 16 + (lh << 3);
  const int ncol  = n0 + wn * 32 + ln;
#pragma unroll
  for (int r = 0; r < 8; ++r) {
    int mg = mbase + r;
    float v0 = acc0[r], v1 = acc1[r];
    if (bias) { v0 += bias[ncol]; v1 += bias[ncol + 16]; }
    if (relu) { v0 = fmaxf(v0, 0.f); v1 = fmaxf(v1, 0.f); }
    size_t i0 = (size_t)mg * N + ncol;
    if (Cres) { Cres[i0] += v0; Cres[i0 + 16] += v1; }
    if (Oh)   { Oh[i0] = (_Float16)v0; Oh[i0 + 16] = (_Float16)v1; }
    if (Of)   { Of[i0] = v0; Of[i0 + 16] = v1; }
  }
}

// ---------------------------------------------------------------------------
// Flash-style causal attention, WMMA, one wave per (b, h, 16-query tile).
// Q/K/V/O: (B*T, D) f16, head h occupies cols [h*HD, (h+1)*HD).
// ---------------------------------------------------------------------------
__global__ __launch_bounds__(32)
void attention_kernel(const _Float16* __restrict__ Q, const _Float16* __restrict__ K,
                      const _Float16* __restrict__ Vv, _Float16* __restrict__ O) {
  const int bh = blockIdx.x;
  const int b  = bh / Hc, hh = bh % Hc;
  const int q0 = blockIdx.y * 16;
  const int lane = threadIdx.x;
  const int lh = lane >> 4, ln = lane & 15;
  const float scale = 0.125f;   // HD^-0.5

  __shared__ alignas(64) _Float16 Plds[16 * 32];
  __shared__ alignas(64) _Float16 Vlds[HDc * 32];   // [dim][key] transposed

  const size_t base = (size_t)b * Tc * Dc + (size_t)hh * HDc;

  // Q tile operands (A layout): lane holds row q0+ln, lane-half selects K range
  const size_t qrow = base + (size_t)(q0 + ln) * Dc;
  v16h aq0 = *(const v16h*)&Q[qrow + lh * 16];        // HD dims 0..31
  v16h aq1 = *(const v16h*)&Q[qrow + 32 + lh * 16];   // HD dims 32..63

  float mrow[8], lrow[8];
  v8f o0 = {}, o1 = {}, o2 = {}, o3 = {};
#pragma unroll
  for (int r = 0; r < 8; ++r) { mrow[r] = -1e30f; lrow[r] = 0.f; }

  for (int kt = 0; kt <= q0; kt += 32) {
    // ---- stage V block (32 keys x 64 dims) transposed into LDS ----
    int key = kt + lane; int kc = key < Tc ? key : Tc - 1;
    size_t vrow = base + (size_t)kc * Dc;
    v16h vv0 = *(const v16h*)&Vv[vrow];
    v16h vv1 = *(const v16h*)&Vv[vrow + 16];
    v16h vv2 = *(const v16h*)&Vv[vrow + 32];
    v16h vv3 = *(const v16h*)&Vv[vrow + 48];
#pragma unroll
    for (int d = 0; d < 16; ++d) {
      Vlds[(d)      * 32 + lane] = vv0[d];
      Vlds[(d + 16) * 32 + lane] = vv1[d];
      Vlds[(d + 32) * 32 + lane] = vv2[d];
      Vlds[(d + 48) * 32 + lane] = vv3[d];
    }

    // ---- K operands for two 16-key score tiles ----
    int key0 = kt + ln;       int kc0 = key0 < Tc ? key0 : Tc - 1;
    int key1 = kt + 16 + ln;  int kc1 = key1 < Tc ? key1 : Tc - 1;
    size_t kr0 = base + (size_t)kc0 * Dc;
    size_t kr1 = base + (size_t)kc1 * Dc;
    v16h bk0lo = *(const v16h*)&K[kr0 + lh * 16];
    v16h bk0hi = *(const v16h*)&K[kr0 + 32 + lh * 16];
    v16h bk1lo = *(const v16h*)&K[kr1 + lh * 16];
    v16h bk1hi = *(const v16h*)&K[kr1 + 32 + lh * 16];

    v8f s0 = {}, s1 = {};
    s0 = wmma_f16(aq0, bk0lo, s0);
    s0 = wmma_f16(aq1, bk0hi, s0);
    s1 = wmma_f16(aq0, bk1lo, s1);
    s1 = wmma_f16(aq1, bk1hi, s1);

    // ---- causal mask + online softmax (rows live in 16-lane halves) ----
#pragma unroll
    for (int r = 0; r < 8; ++r) {
      int m = r + lh * 8;
      int qidx = q0 + m;
      float x0 = s0[r] * scale, x1 = s1[r] * scale;
      if (kt + ln > qidx)      x0 = -1e30f;
      if (kt + 16 + ln > qidx) x1 = -1e30f;
      float bm = fmaxf(x0, x1);
      bm = fmaxf(bm, __shfl_xor(bm, 1, 32));
      bm = fmaxf(bm, __shfl_xor(bm, 2, 32));
      bm = fmaxf(bm, __shfl_xor(bm, 4, 32));
      bm = fmaxf(bm, __shfl_xor(bm, 8, 32));
      float mnew  = fmaxf(mrow[r], bm);
      float alpha = __expf(mrow[r] - mnew);
      float p0 = __expf(x0 - mnew), p1 = __expf(x1 - mnew);
      float rs = p0 + p1;
      rs += __shfl_xor(rs, 1, 32);
      rs += __shfl_xor(rs, 2, 32);
      rs += __shfl_xor(rs, 4, 32);
      rs += __shfl_xor(rs, 8, 32);
      lrow[r] = lrow[r] * alpha + rs;
      mrow[r] = mnew;
      o0[r] *= alpha; o1[r] *= alpha; o2[r] *= alpha; o3[r] *= alpha;
      Plds[m * 32 + ln]      = (_Float16)p0;
      Plds[m * 32 + 16 + ln] = (_Float16)p1;
    }
    __syncthreads();

    // ---- O += P(16x32) @ V(32x64) ----
    v16h ap  = *(const v16h*)&Plds[ln * 32 + lh * 16];
    v16h bv0 = *(const v16h*)&Vlds[(0 * 16 + ln) * 32 + lh * 16];
    v16h bv1 = *(const v16h*)&Vlds[(1 * 16 + ln) * 32 + lh * 16];
    v16h bv2 = *(const v16h*)&Vlds[(2 * 16 + ln) * 32 + lh * 16];
    v16h bv3 = *(const v16h*)&Vlds[(3 * 16 + ln) * 32 + lh * 16];
    o0 = wmma_f16(ap, bv0, o0);
    o1 = wmma_f16(ap, bv1, o1);
    o2 = wmma_f16(ap, bv2, o2);
    o3 = wmma_f16(ap, bv3, o3);
    __syncthreads();
  }

  // ---- normalize and write O (f16, C layout -> global) ----
#pragma unroll
  for (int r = 0; r < 8; ++r) {
    int m = r + lh * 8;
    float inv = 1.f / lrow[r];
    size_t orow = base + (size_t)(q0 + m) * Dc;
    O[orow + 0 * 16 + ln] = (_Float16)(o0[r] * inv);
    O[orow + 1 * 16 + ln] = (_Float16)(o1[r] * inv);
    O[orow + 2 * 16 + ln] = (_Float16)(o2[r] * inv);
    O[orow + 3 * 16 + ln] = (_Float16)(o3[r] * inv);
  }
}

// ---------------------------------------------------------------------------
extern "C" void kernel_launch(void* const* d_in, const int* in_sizes, int n_in,
                              void* d_out, int out_size, void* d_ws, size_t ws_size,
                              hipStream_t stream) {
  const int*   x     = (const int*)  d_in[0];
  const float* tok   = (const float*)d_in[1];
  const float* pos   = (const float*)d_in[2];
  const float* Wq    = (const float*)d_in[3];
  const float* Wk    = (const float*)d_in[4];
  const float* Wv    = (const float*)d_in[5];
  const float* Wproj = (const float*)d_in[6];
  const float* bproj = (const float*)d_in[7];
  const float* ln1g  = (const float*)d_in[8];
  const float* ln1b  = (const float*)d_in[9];
  const float* ln2g  = (const float*)d_in[10];
  const float* ln2b  = (const float*)d_in[11];
  const float* W1    = (const float*)d_in[12];
  const float* b1    = (const float*)d_in[13];
  const float* W2    = (const float*)d_in[14];
  const float* b2    = (const float*)d_in[15];
  const float* lnfg  = (const float*)d_in[16];
  const float* lnfb  = (const float*)d_in[17];
  const float* Whead = (const float*)d_in[18];
  float* out = (float*)d_out;

  // Workspace carve-up (~44 MB total)
  char* w = (char*)d_ws;
  float*    h   = (float*)w;    w += (size_t)Mc * Dc * sizeof(float);
  _Float16* y   = (_Float16*)w; w += (size_t)Mc * Dc * sizeof(_Float16);
  _Float16* qh  = (_Float16*)w; w += (size_t)Mc * Dc * sizeof(_Float16);
  _Float16* kh  = (_Float16*)w; w += (size_t)Mc * Dc * sizeof(_Float16);
  _Float16* vh  = (_Float16*)w; w += (size_t)Mc * Dc * sizeof(_Float16);
  _Float16* ah  = (_Float16*)w; w += (size_t)Mc * Dc * sizeof(_Float16);
  _Float16* ffn = (_Float16*)w; w += (size_t)Mc * DFFc * sizeof(_Float16);
  (void)in_sizes; (void)n_in; (void)out_size; (void)ws_size;

  dim3 blk256(256), blk32(32);
  dim3 gD  (Dc   / TN, Mc / TM);   // N = D
  dim3 gDFF(DFFc / TN, Mc / TM);   // N = DFF
  dim3 gV  (Vc   / TN, Mc / TM);   // N = vocab
  dim3 gAtt(Bc * Hc, Tc / 16);

  embed_kernel<<<Mc, blk256, 0, stream>>>(x, tok, pos, h);

  for (int l = 0; l < Lc; ++l) {
    const size_t DD = (size_t)Dc * Dc;
    layernorm_kernel<<<Mc, blk256, 0, stream>>>(h, ln1g + l * Dc, ln1b + l * Dc, y, Dc);
    gemm_wmma_kernel<<<gD, blk256, 0, stream>>>(y, Wq + l * DD, nullptr,
                                                nullptr, qh, nullptr, Dc, Dc, 0);
    gemm_wmma_kernel<<<gD, blk256, 0, stream>>>(y, Wk + l * DD, nullptr,
                                                nullptr, kh, nullptr, Dc, Dc, 0);
    gemm_wmma_kernel<<<gD, blk256, 0, stream>>>(y, Wv + l * DD, nullptr,
                                                nullptr, vh, nullptr, Dc, Dc, 0);
    attention_kernel<<<gAtt, blk32, 0, stream>>>(qh, kh, vh, ah);
    gemm_wmma_kernel<<<gD, blk256, 0, stream>>>(ah, Wproj + l * DD, bproj + l * Dc,
                                                h, nullptr, nullptr, Dc, Dc, 0);
    layernorm_kernel<<<Mc, blk256, 0, stream>>>(h, ln2g + l * Dc, ln2b + l * Dc, y, Dc);
    gemm_wmma_kernel<<<gDFF, blk256, 0, stream>>>(y, W1 + (size_t)l * Dc * DFFc,
                                                  b1 + (size_t)l * DFFc,
                                                  nullptr, ffn, nullptr, DFFc, Dc, 1);
    gemm_wmma_kernel<<<gD, blk256, 0, stream>>>(ffn, W2 + (size_t)l * DFFc * Dc,
                                                b2 + l * Dc,
                                                h, nullptr, nullptr, Dc, DFFc, 0);
  }

  layernorm_kernel<<<Mc, blk256, 0, stream>>>(h, lnfg, lnfb, y, Dc);
  gemm_wmma_kernel<<<gV, blk256, 0, stream>>>(y, Whead, nullptr,
                                              nullptr, nullptr, out, Vc, Dc, 0);
}
